// GraphAttentionLayer_P1_35399120453962
// MI455X (gfx1250) — matrix operand order repaired and checked by reading, not compiled, not run on hardware
//
#include <hip/hip_runtime.h>
#include <hip/hip_bf16.h>
#include <math.h>

// ---------------------------------------------------------------------------
// GAT layer, fused for MI455X (gfx1250).
//   N=8192, IN_C=128, OUT_C=64, ALPHA=0.5
// Roofline: adj (256 MB int32) read exactly once, non-temporal, @23.3 TB/s
// ~= 11 us floor. The 8192^2 attention matrix is never materialized; softmax
// is single-pass using the global-f2-max upper bound (leaky_relu is monotone,
// so lrelu(f1[i]+max_j f2[j]) >= true row max; exp args stay <= 0).
// Matrix work: h@W on v_wmma_f32_16x16x4_f32 (exact), attention@Wh on
// v_wmma_f32_16x16x32_bf16 (f32 accumulate). Workspace: ~12.7 MB.
// ---------------------------------------------------------------------------

#define NN    8192
#define INC   128
#define OUTC  64
#define ALPHA 0.5f
#define SEGS  4
#define SEGW  (NN / SEGS)   // 2048 columns per segment

typedef float  v2f   __attribute__((ext_vector_type(2)));
typedef float  v4f   __attribute__((ext_vector_type(4)));
typedef float  v8f   __attribute__((ext_vector_type(8)));
typedef int    v4i   __attribute__((ext_vector_type(4)));
typedef __bf16 v4bf  __attribute__((ext_vector_type(4)));
typedef __bf16 v8bf  __attribute__((ext_vector_type(8)));
typedef __bf16 v16bf __attribute__((ext_vector_type(16)));

__device__ __forceinline__ float lrelu(float v) { return fmaxf(v, ALPHA * v); }

// ---------------------------------------------------------------------------
// Kernel 1: Wh = h @ W   (8192x128 @ 128x64), exact f32 via WMMA 16x16x4.
// 4 waves/block, one 16x16 n-tile per wave, 32 WMMA steps over K=128.
// Also writes bf16 copy of Wh for the attention GEMM's B operand.
// ---------------------------------------------------------------------------
__global__ void gat_wh_gemm(const float* __restrict__ h,
                            const float* __restrict__ W,
                            float* __restrict__ Wh,
                            __bf16* __restrict__ WhB) {
  const int lane  = threadIdx.x & 31;
  const int wv    = threadIdx.x >> 5;     // n-tile 0..3
  const int rbase = blockIdx.x * 16;
  const int n0    = wv * 16;
  const int m     = lane & 15;            // A: row,  B: col
  const int half  = lane >> 4;            // K-half selector

  v8f c = {};
  const float* hrow = h + (size_t)(rbase + m) * INC;
  #pragma unroll 4
  for (int k0 = 0; k0 < INC; k0 += 4) {
    // ISA 32-bit A 16x4 layout: lanes 0-15 K={0,1}, lanes 16-31 K={2,3}
    v2f a, b;
    a.x = hrow[k0 + 2 * half + 0];
    a.y = hrow[k0 + 2 * half + 1];
    b.x = W[(k0 + 2 * half + 0) * OUTC + n0 + m];
    b.y = W[(k0 + 2 * half + 1) * OUTC + n0 + m];
    c = __builtin_amdgcn_wmma_f32_16x16x4_f32(false, a, false, b,
                                              (short)0, c, false, false);
  }
  // C layout: lane<16 -> M=r, lane>=16 -> M=r+8; N = lane&15
  #pragma unroll
  for (int r = 0; r < 8; ++r) {
    const int row = rbase + r + half * 8;
    const int col = n0 + m;
    Wh [(size_t)row * OUTC + col] = c[r];
    WhB[(size_t)row * OUTC + col] = (__bf16)c[r];
  }
}

// ---------------------------------------------------------------------------
// Kernel 2: f1 = Wh@a1, f2 = Wh@a2  (one row per thread)
// ---------------------------------------------------------------------------
__global__ void gat_f1f2(const float* __restrict__ Wh,
                         const float* __restrict__ a,
                         float* __restrict__ f1, float* __restrict__ f2) {
  const int row = blockIdx.x * blockDim.x + threadIdx.x;
  if (row >= NN) return;
  const float* wr = Wh + (size_t)row * OUTC;
  float s1 = 0.f, s2 = 0.f;
  #pragma unroll 8
  for (int k = 0; k < OUTC; ++k) {
    const float w = wr[k];
    s1 += w * a[k];
    s2 += w * a[OUTC + k];
  }
  f1[row] = s1;
  f2[row] = s2;
}

// ---------------------------------------------------------------------------
// Kernel 3: global max of f2 (enables single-pass softmax)
// ---------------------------------------------------------------------------
__global__ void gat_f2max(const float* __restrict__ f2, float* __restrict__ gmax) {
  __shared__ float red[256];
  float m = -3.0e38f;
  for (int i = threadIdx.x; i < NN; i += 256) m = fmaxf(m, f2[i]);
  red[threadIdx.x] = m;
  __syncthreads();
  for (int s = 128; s > 0; s >>= 1) {
    if ((int)threadIdx.x < s)
      red[threadIdx.x] = fmaxf(red[threadIdx.x], red[threadIdx.x + s]);
    __syncthreads();
  }
  if (threadIdx.x == 0) *gmax = red[0];
}

// ---------------------------------------------------------------------------
// Kernel 4: fused masked-softmax x Wh, flash style. One wave owns 16 rows and
// a 2048-column segment, processed 128 columns per outer iteration:
//   - 16x b128 non-temporal adj loads (one int4 per lane per row)
//   - 64 exp weights per lane -> packed ds_store_b64 bf16 p-tile (16x128)
//   - 4 sub-chunks of K=32: A frag from LDS (2x ds_load_b128, ISA 16-bit A
//     layout), 4 B n-tiles from bf16 Wh, 5x v_wmma_f32_16x16x32_bf16
//     (4 output tiles + ones-B tile accumulating softmax row sums for free).
// ---------------------------------------------------------------------------
__global__ void gat_attn(const int* __restrict__ adj,
                         const __bf16* __restrict__ WhB,
                         const float* __restrict__ f1,
                         const float* __restrict__ f2,
                         const float* __restrict__ gmaxp,
                         float* __restrict__ accP,
                         float* __restrict__ sumP) {
  __shared__ __align__(16) __bf16 ptile[4][16 * 128];  // per-wave 16x128 tile

  const int lane   = threadIdx.x & 31;
  const int wv     = threadIdx.x >> 5;
  const int gw     = blockIdx.x * 4 + wv;   // global wave id, 0..2047
  const int rowblk = gw >> 2;               // 0..511
  const int seg    = gw & 3;                // 0..3
  const int rbase  = rowblk * 16;
  const int jbeg   = seg * SEGW;
  const int jend   = jbeg + SEGW;
  const float gmax = *gmaxp;

  // Per-row logit offsets and softmax upper bound
  float f1r[16], mrow[16];
  #pragma unroll
  for (int m = 0; m < 16; ++m) {
    f1r[m]  = f1[rbase + m];
    mrow[m] = lrelu(f1r[m] + gmax);
  }

  v16bf onesB;
  #pragma unroll
  for (int t = 0; t < 16; ++t) onesB[t] = (__bf16)1.0f;

  v8f acc0 = {}, acc1 = {}, acc2 = {}, acc3 = {}, accS = {};
  __bf16* tile = ptile[wv];
  const int half = lane >> 4;
  const int mA   = lane & 15;

  for (int j0 = jbeg; j0 < jend; j0 += 128) {
    const size_t base = (size_t)rbase * NN + j0 + lane * 4;
    const v4f f2v = *(const v4f*)(f2 + j0 + lane * 4);

    // Cooperative p-tile: lane covers 4 consecutive columns, loop 16 rows.
    // adj is stream-once -> non-temporal so it never evicts Wh from L2.
    #pragma unroll
    for (int m = 0; m < 16; ++m) {
      const v4i av = __builtin_nontemporal_load((const v4i*)(adj + base + (size_t)m * NN));
      v4bf pv;
      #pragma unroll
      for (int q = 0; q < 4; ++q) {
        const float s = lrelu(f1r[m] + f2v[q]);
        const float p = (av[q] > 0) ? __expf(s - mrow[m]) : 0.0f;  // in (0,1]
        pv[q] = (__bf16)p;
      }
      *(v4bf*)(tile + m * 128 + lane * 4) = pv;   // packed ds_store_b64
    }
    if (j0 + 128 < jend) __builtin_prefetch(&adj[base + 128], 0, 1);

    #pragma unroll
    for (int sc = 0; sc < 4; ++sc) {
      // A fragment, ISA 16-bit 16x32 layout:
      // lane<16: K={0..7,16..23}; lane>=16: K={8..15,24..31}
      const __bf16* arow = tile + mA * 128 + sc * 32 + half * 8;
      const v8bf alo = *(const v8bf*)(arow);
      const v8bf ahi = *(const v8bf*)(arow + 16);
      v16bf A;
      #pragma unroll
      for (int t = 0; t < 8; ++t) { A[t] = alo[t]; A[t + 8] = ahi[t]; }

      // B fragments: col n = lane&15, K order same as A (lane<16: K={0..7,16..23}; lane>=16: K={8..15,24..31}), 4 n-tiles of 16.
      const __bf16* wcol = WhB + (size_t)(j0 + sc * 32 + half * 8) * OUTC + mA;
      v16bf B0, B1, B2, B3;
      #pragma unroll
      for (int t = 0; t < 8; ++t) {
        B0[t] = wcol[t * OUTC +  0]; B0[t + 8] = wcol[(t + 16) * OUTC +  0];
        B1[t] = wcol[t * OUTC + 16]; B1[t + 8] = wcol[(t + 16) * OUTC + 16];
        B2[t] = wcol[t * OUTC + 32]; B2[t + 8] = wcol[(t + 16) * OUTC + 32];
        B3[t] = wcol[t * OUTC + 48]; B3[t + 8] = wcol[(t + 16) * OUTC + 48];
      }

      acc0 = __builtin_amdgcn_wmma_f32_16x16x32_bf16(false, A, false, B0, (short)0, acc0, false, false);
      acc1 = __builtin_amdgcn_wmma_f32_16x16x32_bf16(false, A, false, B1, (short)0, acc1, false, false);
      acc2 = __builtin_amdgcn_wmma_f32_16x16x32_bf16(false, A, false, B2, (short)0, acc2, false, false);
      acc3 = __builtin_amdgcn_wmma_f32_16x16x32_bf16(false, A, false, B3, (short)0, acc3, false, false);
      accS = __builtin_amdgcn_wmma_f32_16x16x32_bf16(false, A, false, onesB, (short)0, accS, false, false);
    }
  }

  // Epilogue: accS's every column holds the row sum -> same (lane,r) slot as
  // the output tiles, so normalization data is already in-lane.
  const int col = lane & 15;
  #pragma unroll
  for (int r = 0; r < 8; ++r) {
    const int    m   = r + half * 8;
    const size_t row = (size_t)seg * NN + (rbase + m);
    float* orow = accP + row * OUTC;
    orow[ 0 + col] = acc0[r];
    orow[16 + col] = acc1[r];
    orow[32 + col] = acc2[r];
    orow[48 + col] = acc3[r];
    if (col == 0) sumP[row] = accS[r];
  }
}

// ---------------------------------------------------------------------------
// Kernel 5: merge the 4 column-segment partials, normalize, leaky-relu.
// ---------------------------------------------------------------------------
__global__ void gat_combine(const float* __restrict__ accP,
                            const float* __restrict__ sumP,
                            float* __restrict__ out) {
  const int idx = blockIdx.x * blockDim.x + threadIdx.x;
  if (idx >= NN * OUTC) return;
  const int i = idx >> 6;
  const int n = idx & 63;
  float s = 0.f, v = 0.f;
  #pragma unroll
  for (int g = 0; g < SEGS; ++g) {
    s += sumP[(size_t)g * NN + i];
    v += accP[((size_t)g * NN + i) * OUTC + n];
  }
  out[idx] = lrelu(v / s);
}

// ---------------------------------------------------------------------------
extern "C" void kernel_launch(void* const* d_in, const int* in_sizes, int n_in,
                              void* d_out, int out_size, void* d_ws, size_t ws_size,
                              hipStream_t stream) {
  const float* h   = (const float*)d_in[0];   // (8192,128) f32
  const int*   adj = (const int*)  d_in[1];   // (8192,8192) i32
  const float* W   = (const float*)d_in[2];   // (128,64) f32
  const float* a   = (const float*)d_in[3];   // (128,1) f32
  float* out = (float*)d_out;                 // (8192,64) f32

  // Workspace layout (~12.7 MB total):
  char* ws = (char*)d_ws;
  float*  Wh   = (float*) (ws);                                   // 2 MB
  __bf16* WhB  = (__bf16*)(ws + (size_t)2 * 1024 * 1024);         // 1 MB
  float*  f1   = (float*) (ws + (size_t)3 * 1024 * 1024);         // 32 KB
  float*  f2   = (float*) (ws + (size_t)3 * 1024 * 1024 + 32768); // 32 KB
  float*  gmax = (float*) (ws + (size_t)3 * 1024 * 1024 + 65536); // 4 B
  float*  accP = (float*) (ws + (size_t)4 * 1024 * 1024);         // 8 MB
  float*  sumP = (float*) (ws + (size_t)12 * 1024 * 1024);        // 128 KB

  gat_wh_gemm<<<NN / 16, 128, 0, stream>>>(h, W, Wh, WhB);
  gat_f1f2   <<<NN / 256, 256, 0, stream>>>(Wh, a, f1, f2);
  gat_f2max  <<<1, 256, 0, stream>>>(f2, gmax);
  gat_attn   <<<(NN / 16) * SEGS / 4, 128, 0, stream>>>(adj, WhB, f1, f2, gmax, accP, sumP);
  gat_combine<<<(NN * OUTC) / 256, 256, 0, stream>>>(accP, sumP, out);
}
